// BiLinearAttention_19456201851322
// MI455X (gfx1250) — compile-verified, matrix operand order checked by reading, not run on hardware
//
#include <hip/hip_runtime.h>

// ---------------------------------------------------------------------------
// BiLinearAttention on MI455X (gfx1250):
//   GEMM1: encW[e,g] = sum_h enc[e,h] * W[h,g]          (16384 x 512 x 512)
//   GEMM2: scores[b,d,e] = sum_g dec[b,d,g]*encW[b,e,g] + bias  (8 batches, 2048x2048x512)
// Compute-bound (≈220 FLOP/byte) -> run on V_WMMA_F32_16X16X32_BF16.
// ---------------------------------------------------------------------------

typedef __attribute__((ext_vector_type(16))) __bf16 v16bf;
typedef __attribute__((ext_vector_type(8)))  float  v8f;

union Frag {
    uint4 u[2];   // two b128 loads fill the 8-VGPR 16x32 bf16 fragment
    v16bf v;
};

__device__ __forceinline__ unsigned short f2bf(float f) {
    // round-to-nearest-even fp32 -> bf16
    unsigned u = __builtin_bit_cast(unsigned, f);
    u += 0x7FFFu + ((u >> 16) & 1u);
    return (unsigned short)(u >> 16);
}

// ---------------------------------------------------------------------------
// Elementwise fp32 -> bf16 conversion, 4 elements / thread (b128 in, b64 out)
// ---------------------------------------------------------------------------
__global__ __launch_bounds__(256) void cvt_f32_to_bf16_vec4(
    const float4* __restrict__ in, ushort4* __restrict__ out, int n4)
{
    int i = blockIdx.x * blockDim.x + threadIdx.x;
    if (i >= n4) return;
    float4 f = in[i];
    ushort4 o;
    o.x = f2bf(f.x); o.y = f2bf(f.y); o.z = f2bf(f.z); o.w = f2bf(f.w);
    out[i] = o;
}

// ---------------------------------------------------------------------------
// W [512 x 512] fp32 row-major -> Wt [512 x 512] bf16 (transposed, K-contig)
// ---------------------------------------------------------------------------
__global__ __launch_bounds__(256) void cvt_transpose_512(
    const float* __restrict__ W, unsigned short* __restrict__ Wt)
{
    int idx = blockIdx.x * blockDim.x + threadIdx.x; // idx = h*512 + g
    int h = idx >> 9;
    int g = idx & 511;
    Wt[g * 512 + h] = f2bf(W[idx]);
}

// ---------------------------------------------------------------------------
// C = A * Bt^T  (A: MxK row-major bf16, Bt: NxK row-major bf16), batched.
// Block: 256 threads = 8 waves; wave tile 64x64 (4x4 WMMA accums);
// block tile 128(M) x 256(N). All dims divide tiles exactly -> no bounds.
// STORE_BF16: write bf16 (GEMM1 intermediate) else fp32 (+bias) (GEMM2).
// ---------------------------------------------------------------------------
template <bool STORE_BF16>
__global__ __launch_bounds__(256) void gemm_tn_bf16_wmma(
    const unsigned short* __restrict__ A,
    const unsigned short* __restrict__ Bt,
    void* __restrict__ Cout,
    int N, int K,
    long long strideA, long long strideB, long long strideC,
    const float* __restrict__ biasPtr)
{
    const int bat = blockIdx.z;
    const unsigned short* __restrict__ Ab = A  + (size_t)bat * strideA;
    const unsigned short* __restrict__ Bb = Bt + (size_t)bat * strideB;

    const int lane = threadIdx.x & 31;
    const int wave = threadIdx.x >> 5;
    const int wm   = wave & 1;      // 2 waves along M
    const int wn   = wave >> 1;     // 4 waves along N
    const int row0 = blockIdx.y * 128 + wm * 64;
    const int col0 = blockIdx.x * 256 + wn * 64;

    const int ml = lane & 15;       // row (A) / col (B) within 16-tile
    const int hi = lane >> 4;       // selects K-halves per WMMA fragment layout

    v8f acc[4][4];
#pragma unroll
    for (int mi = 0; mi < 4; ++mi)
#pragma unroll
        for (int ni = 0; ni < 4; ++ni)
            acc[mi][ni] = (v8f)0.0f;

    for (int k0 = 0; k0 < K; k0 += 32) {
        Frag a[4], b[4];
#pragma unroll
        for (int mi = 0; mi < 4; ++mi) {
            const unsigned short* p =
                Ab + (size_t)(row0 + mi * 16 + ml) * K + k0 + hi * 8;
            a[mi].u[0] = *(const uint4*)(p);        // K = k0+hi*8   .. +8
            a[mi].u[1] = *(const uint4*)(p + 16);   // K = k0+16+hi*8.. +8
        }
#pragma unroll
        for (int ni = 0; ni < 4; ++ni) {
            const unsigned short* p =
                Bb + (size_t)(col0 + ni * 16 + ml) * K + k0 + hi * 8;
            b[ni].u[0] = *(const uint4*)(p);
            b[ni].u[1] = *(const uint4*)(p + 16);
        }
        // hint the next K-slab toward the caches (global_prefetch_b8)
        if (k0 + 32 < K) {
            __builtin_prefetch(Ab + (size_t)(row0 + ml) * K + k0 + 32, 0, 1);
            __builtin_prefetch(Bb + (size_t)(col0 + ml) * K + k0 + 32, 0, 1);
        }
#pragma unroll
        for (int mi = 0; mi < 4; ++mi)
#pragma unroll
            for (int ni = 0; ni < 4; ++ni)
                acc[mi][ni] = __builtin_amdgcn_wmma_f32_16x16x32_bf16(
                    /*neg_a=*/false, a[mi].v,
                    /*neg_b=*/false, b[ni].v,
                    /*c_mod=*/(short)0, acc[mi][ni],
                    /*reuse_a=*/false, /*reuse_b=*/false);
    }

    const float bias = (biasPtr != nullptr) ? biasPtr[0] : 0.0f;

#pragma unroll
    for (int mi = 0; mi < 4; ++mi) {
#pragma unroll
        for (int ni = 0; ni < 4; ++ni) {
            const int col = col0 + ni * 16 + ml;
#pragma unroll
            for (int r = 0; r < 8; ++r) {
                const int row = row0 + mi * 16 + hi * 8 + r;
                const size_t idx =
                    (size_t)bat * strideC + (size_t)row * N + col;
                const float val = acc[mi][ni][r] + bias;
                if constexpr (STORE_BF16) {
                    ((unsigned short*)Cout)[idx] = f2bf(val);
                } else {
                    ((float*)Cout)[idx] = val;
                }
            }
        }
    }
}

// ---------------------------------------------------------------------------
// Host-side launch
// ---------------------------------------------------------------------------
extern "C" void kernel_launch(void* const* d_in, const int* in_sizes, int n_in,
                              void* d_out, int out_size, void* d_ws, size_t ws_size,
                              hipStream_t stream) {
    const float* enc = (const float*)d_in[0];  // [8, 2048, 512]
    const float* dec = (const float*)d_in[1];  // [8, 2048, 512]
    const float* W   = (const float*)d_in[2];  // [1, 512, 512]
    const float* b   = (const float*)d_in[3];  // [1]

    constexpr int    Bn = 8, S = 2048, H = 512;
    constexpr size_t encElems = (size_t)Bn * S * H;   // 8,388,608

    // workspace layout (bf16, ~51 MB total)
    unsigned short* enc_bf  = (unsigned short*)d_ws;
    unsigned short* dec_bf  = enc_bf + encElems;
    unsigned short* Wt_bf   = dec_bf + encElems;
    unsigned short* encW_bf = Wt_bf + (size_t)H * H;

    const int n4 = (int)(encElems / 4);               // 2,097,152
    cvt_f32_to_bf16_vec4<<<n4 / 256, 256, 0, stream>>>(
        (const float4*)enc, (ushort4*)enc_bf, n4);
    cvt_f32_to_bf16_vec4<<<n4 / 256, 256, 0, stream>>>(
        (const float4*)dec, (ushort4*)dec_bf, n4);
    cvt_transpose_512<<<(H * H) / 256, 256, 0, stream>>>(W, Wt_bf);

    // GEMM1: encW[B*S, H] = enc[B*S, H] @ Wt[H, H]^T   (bf16 out)
    {
        dim3 grid(H / 256, (Bn * S) / 128, 1);
        gemm_tn_bf16_wmma<true><<<grid, 256, 0, stream>>>(
            enc_bf, Wt_bf, (void*)encW_bf,
            /*N=*/H, /*K=*/H,
            /*strideA=*/0, /*strideB=*/0, /*strideC=*/0,
            /*bias=*/nullptr);
    }

    // GEMM2: scores[b] = dec[b] @ encW[b]^T + bias     (fp32 out)
    {
        dim3 grid(S / 256, S / 128, Bn);
        gemm_tn_bf16_wmma<false><<<grid, 256, 0, stream>>>(
            dec_bf, encW_bf, d_out,
            /*N=*/S, /*K=*/H,
            /*strideA=*/(long long)S * H,
            /*strideB=*/(long long)S * H,
            /*strideC=*/(long long)S * S,
            /*bias=*/b);
    }
}
